// MultiAxisAttention_71768903517014
// MI455X (gfx1250) — matrix operand order repaired and checked by reading, not compiled, not run on hardware
//
#include <hip/hip_runtime.h>
#include <cstdint>
#include <cstddef>

typedef __attribute__((ext_vector_type(16))) _Float16 v16h;
typedef __attribute__((ext_vector_type(8)))  _Float16 f16x8;
typedef __attribute__((ext_vector_type(8)))  float    v8f;

#define NTOK   343      // tokens per window (7^3)
#define NPAD   352      // padded to 22 tiles of 16
#define DIMD   128
#define HEADS_ 4
#define DH     32
#define NWIN   128      // 2*4*4*4 windows
#define QK_SCALE 0.17677669529663687f   // 32^-0.5

// ---- workspace layout (bytes); total ~44.8 MB ----
static constexpr size_t OFF_WQKVT = 0;                        // 384*128 f16
static constexpr size_t OFF_WOUTT = 98304;                    // 128*128 f16
static constexpr size_t OFF_BIAS  = 131072;                   // 4*343*343 f32
static constexpr size_t OFF_Q     = 2013696;                  // 128*4*343*32 f16
static constexpr size_t OFF_K     = OFF_Q + 11239424ull;
static constexpr size_t OFF_V     = OFF_K + 11239424ull;
static constexpr size_t OFF_CTX   = OFF_V + 11239424ull;      // 128*343*128 f16

// ---------------- helpers ----------------
__device__ __forceinline__ v16h ld16(const _Float16* p0, const _Float16* p1) {
  f16x8 a = *(const f16x8*)p0;
  f16x8 b = *(const f16x8*)p1;
  return __builtin_shufflevector(a, b, 0,1,2,3,4,5,6,7,8,9,10,11,12,13,14,15);
}
__device__ __forceinline__ v8f wmma32(v16h a, v16h b, v8f c) {
  return __builtin_amdgcn_wmma_f32_16x16x32_f16(false, a, false, b, (short)0, c, false, false);
}
__device__ __forceinline__ void wait_lds() {
  __asm__ volatile("s_wait_dscnt 0" ::: "memory");
}
__device__ __forceinline__ void wait_async() {
  __asm__ volatile("s_wait_asynccnt 0x0" ::: "memory");
}
// bulk contiguous global -> LDS copy on the async path (ASYNCcnt-tracked)
__device__ __forceinline__ void async_copy_b128(_Float16* lds_dst, const _Float16* gsrc) {
  unsigned loff = (unsigned)(unsigned long long)lds_dst;   // generic LDS addr[31:0] = LDS offset
  __asm__ volatile("global_load_async_to_lds_b128 %0, %1, off"
                   :: "v"(loff), "v"(gsrc) : "memory");
}

// ---------------- kernel 0a: transpose+f16 weights ----------------
__global__ void prep_weights(const float* __restrict__ wqkv, const float* __restrict__ wout,
                             _Float16* __restrict__ wqkvT, _Float16* __restrict__ woutT) {
  int t = blockIdx.x * 256 + threadIdx.x;
  if (t < 384 * 128) {
    int col = t / 128, k = t % 128;               // wqkvT[col][k]
    wqkvT[t] = (_Float16)wqkv[k * 384 + col];
  } else {
    int t2 = t - 384 * 128;
    if (t2 < 128 * 128) {
      int col = t2 / 128, k = t2 % 128;           // woutT[col][k]
      woutT[t2] = (_Float16)wout[k * 128 + col];
    }
  }
}

// ---------------- kernel 0b: bias[h][i][j] gather ----------------
__global__ void prep_bias(const float* __restrict__ table, const int* __restrict__ rel,
                          float* __restrict__ biasm) {
  int t = blockIdx.x * 256 + threadIdx.x;
  const int NN = NTOK * NTOK;
  if (t < HEADS_ * NN) {
    int h = t / NN;
    int rem = t % NN;
    biasm[t] = table[rel[rem] * HEADS_ + h];
  }
}

// ---------------- kernel A: LayerNorm + QKV GEMM ----------------
// grid = (NWIN, 3); each wave owns one 16-row m-tile.
__global__ void __launch_bounds__(256) ln_qkv(const float* __restrict__ x,
                                              const float* __restrict__ lnw,
                                              const float* __restrict__ lnb,
                                              const _Float16* __restrict__ wqkvT,
                                              _Float16* __restrict__ qws,
                                              _Float16* __restrict__ kws,
                                              _Float16* __restrict__ vws) {
  __shared__ _Float16 xs[8 * 16 * DIMD];   // xn slab per wave
  __shared__ _Float16 st[8 * 16 * 16];     // D-tile staging per wave
  int w = blockIdx.x;
  int tid = threadIdx.x;
  int wave = tid >> 5, lane = tid & 31, lane16 = lane & 15, half = lane >> 4;
  int mt = wave + 8 * (int)blockIdx.y;
  if (mt >= 22) return;                    // no block-wide barriers in this kernel
  _Float16* slab = xs + wave * 16 * DIMD;
  _Float16* stg  = st + wave * 256;

  // --- LayerNorm: 2 lanes per row, 64 elems each ---
  int lrow = lane >> 1;
  int hsel = lane & 1;
  int row = mt * 16 + lrow;
  const float* xp = x + ((size_t)w * NTOK + row) * DIMD + hsel * 64;
  float s = 0.f, ss = 0.f;
  if (row < NTOK) {
    for (int e = 0; e < 64; e += 4) {
      float4 v = *(const float4*)(xp + e);
      s  += v.x + v.y + v.z + v.w;
      ss += v.x*v.x + v.y*v.y + v.z*v.z + v.w*v.w;
    }
  }
  s  += __shfl_xor(s, 1, 32);
  ss += __shfl_xor(ss, 1, 32);
  float mu   = s * (1.f / 128.f);
  float var  = ss * (1.f / 128.f) - mu * mu;
  float rstd = rsqrtf(var + 1e-5f);
  _Float16* dst = slab + lrow * DIMD + hsel * 64;
  if (row < NTOK) {
    for (int e = 0; e < 64; ++e) {
      int g = hsel * 64 + e;
      dst[e] = (_Float16)(((xp[e] - mu) * rstd) * lnw[g] + lnb[g]);
    }
  } else {
    for (int e = 0; e < 64; ++e) dst[e] = (_Float16)0.f;
  }
  wait_lds();

  // --- A fragments: invariant across the 24 n-tiles ---
  v16h afrag[4];
  for (int kc = 0; kc < 4; ++kc) {
    const _Float16* ap = slab + lane16 * DIMD + kc * 32 + half * 16;
    afrag[kc] = ld16(ap, ap + 8);
  }

  int rlocal = lane >> 1, seg = lane & 1;
  int rr = mt * 16 + rlocal;
  for (int nt = 0; nt < 24; ++nt) {
    v8f acc = {};
    const _Float16* bcol = wqkvT + (nt * 16 + lane16) * DIMD;
    for (int kc = 0; kc < 4; ++kc) {
      const _Float16* bp = bcol + kc * 32 + half * 16;
      acc = wmma32(afrag[kc], ld16(bp, bp + 8), acc);
    }
    int col0 = nt * 16;
    int part = col0 >> 7;                 // 0=q 1=k 2=v (uniform)
    float fac = (part == 0) ? QK_SCALE : 1.f;
    int hh = (col0 >> 5) & 3;
    int doff = col0 & 16;                 // 0 or 16 within head dim
    // stage D tile [16x16] f16 in LDS
    for (int r = 0; r < 8; ++r)
      stg[(r + half * 8) * 16 + lane16] = (_Float16)(acc[r] * fac);
    wait_lds();
    // vectorized copy-out: 2 lanes per row, one b128 store each
    f16x8 dvec = *(const f16x8*)(stg + rlocal * 16 + seg * 8);
    _Float16* gbase = (part == 0 ? qws : (part == 1 ? kws : vws));
    if (rr < NTOK) {
      *(f16x8*)(gbase + ((size_t)(w * HEADS_ + hh) * NTOK + rr) * DH + doff + seg * 8) = dvec;
    }
  }
}

// ---------------- kernel B: flash attention per (window, head) ----------------
__global__ void __launch_bounds__(256) attn(const _Float16* __restrict__ qws,
                                            const _Float16* __restrict__ kws,
                                            const _Float16* __restrict__ vws,
                                            const float* __restrict__ biasm,
                                            _Float16* __restrict__ ctx) {
  __shared__ _Float16 ks[NPAD * DH];       // K row-major [key][dh]
  __shared__ _Float16 vts[DH * NPAD];      // V transposed [dh][key]
  __shared__ _Float16 ps[8 * 16 * 32];     // per-wave P / output staging
  int w = blockIdx.x >> 2;
  int h = blockIdx.x & 3;
  int tid = threadIdx.x;
  int wave = tid >> 5, lane = tid & 31, lane16 = lane & 15, half = lane >> 4;

  const _Float16* qh = qws + (size_t)(w * HEADS_ + h) * NTOK * DH;
  const _Float16* kh = kws + (size_t)(w * HEADS_ + h) * NTOK * DH;
  const _Float16* vh = vws + (size_t)(w * HEADS_ + h) * NTOK * DH;
  const float*    bh = biasm + (size_t)h * NTOK * NTOK;

  // zero the padded K rows (343..351)
  for (int i = tid; i < (NPAD - NTOK) * DH; i += 256)
    ks[NTOK * DH + i] = (_Float16)0.f;
  // async bulk copy of K (343*64B) straight into LDS
  for (int ci = tid; ci < NTOK * DH / 8; ci += 256)
    async_copy_b128(ks + ci * 8, kh + ci * 8);
  // V transpose: vectorized global read, 2B LDS scatter
  for (int ci = tid; ci < NPAD * DH / 8; ci += 256) {
    int vrow = ci >> 2, ds = (ci & 3) * 8;
    f16x8 v = {};
    if (vrow < NTOK) v = *(const f16x8*)(vh + vrow * DH + ds);
    for (int e = 0; e < 8; ++e) vts[(ds + e) * NPAD + vrow] = v[e];
  }
  wait_async();
  __syncthreads();

  _Float16* pb = ps + wave * 16 * 32;
  int rlocal = lane >> 1, seg = lane & 1;
  for (int qt = wave; qt < 22; qt += 8) {
    int arow = qt * 16 + lane16;
    v16h aq = {};
    if (arow < NTOK) {
      const _Float16* qp = qh + (size_t)arow * DH + half * 16;
      aq = ld16(qp, qp + 8);
    }
    // clamped per-row bias pointers (hoisted; padded rows read row 342, never stored)
    const float* brow[8];
    for (int r = 0; r < 8; ++r) {
      int rowq = qt * 16 + r + half * 8;
      int rc = rowq < NTOK ? rowq : NTOK - 1;
      brow[r] = bh + (size_t)rc * NTOK;
    }
    float mrow[8], srow[8];
    v8f o0 = {}, o1 = {};
    for (int r = 0; r < 8; ++r) { mrow[r] = -1e30f; srow[r] = 0.f; }

    for (int c = 0; c < 11; ++c) {          // 11 chunks of 32 keys (343 -> 352)
      int key0 = c * 32 + lane16;           // always < 343
      int key1 = key0 + 16;                 // invalid only in chunk 10
      int key1c = key1 < NTOK ? key1 : NTOK - 1;
      const _Float16* b0 = ks + key0 * DH + half * 16;
      const _Float16* b1 = ks + (key0 + 16) * DH + half * 16;
      v8f zz = {};
      v8f s0 = wmma32(aq, ld16(b0, b0 + 8), zz);
      v8f s1 = wmma32(aq, ld16(b1, b1 + 8), zz);

      for (int r = 0; r < 8; ++r) {
        float x0 = s0[r] + brow[r][key0];
        float x1 = s1[r] + brow[r][key1c];
        x1 = (key1 < NTOK) ? x1 : -1e30f;   // mask padded keys, branch-free
        float mx = fmaxf(x0, x1);
        mx = fmaxf(mx, __shfl_xor(mx, 1, 32));
        mx = fmaxf(mx, __shfl_xor(mx, 2, 32));
        mx = fmaxf(mx, __shfl_xor(mx, 4, 32));
        mx = fmaxf(mx, __shfl_xor(mx, 8, 32));
        float mnew = fmaxf(mrow[r], mx);
        float cf = __expf(mrow[r] - mnew);
        float p0 = __expf(x0 - mnew);
        float p1 = __expf(x1 - mnew);
        float rs = p0 + p1;
        rs += __shfl_xor(rs, 1, 32);
        rs += __shfl_xor(rs, 2, 32);
        rs += __shfl_xor(rs, 4, 32);
        rs += __shfl_xor(rs, 8, 32);
        srow[r] = srow[r] * cf + rs;
        mrow[r] = mnew;
        o0[r] *= cf;
        o1[r] *= cf;
        pb[(r + half * 8) * 32 + lane16]      = (_Float16)p0;
        pb[(r + half * 8) * 32 + 16 + lane16] = (_Float16)p1;
      }
      wait_lds();   // cross-lane LDS P-store -> A-load ordering
      const _Float16* app = pb + lane16 * 32 + half * 16;
      v16h ap = ld16(app, app + 8);
      const _Float16* bv0 = vts + lane16 * NPAD + c * 32 + half * 16;
      const _Float16* bv1 = vts + (16 + lane16) * NPAD + c * 32 + half * 16;
      o0 = wmma32(ap, ld16(bv0, bv0 + 8), o0);
      o1 = wmma32(ap, ld16(bv1, bv1 + 8), o1);
    }

    // epilogue: normalize, stage 16x32 f16 tile in LDS, copy out with b128 stores
    for (int r = 0; r < 8; ++r) {
      float inv = srow[r] > 0.f ? 1.f / srow[r] : 0.f;
      pb[(r + half * 8) * 32 + lane16]      = (_Float16)(o0[r] * inv);
      pb[(r + half * 8) * 32 + 16 + lane16] = (_Float16)(o1[r] * inv);
    }
    wait_lds();
    int rr = qt * 16 + rlocal;
    if (rr < NTOK) {
      f16x8 d0 = *(const f16x8*)(pb + rlocal * 32 + seg * 16);
      f16x8 d1 = *(const f16x8*)(pb + rlocal * 32 + seg * 16 + 8);
      _Float16* cp = ctx + ((size_t)w * NTOK + rr) * DIMD + h * DH + seg * 16;
      *(f16x8*)cp = d0;
      *(f16x8*)(cp + 8) = d1;
    }
  }
}

// ---------------- kernel C: output projection GEMM ----------------
__global__ void __launch_bounds__(256) proj(const _Float16* __restrict__ ctx,
                                            const _Float16* __restrict__ woutT,
                                            float* __restrict__ out) {
  int tid = threadIdx.x;
  int wave = tid >> 5, lane = tid & 31, lane16 = lane & 15, half = lane >> 4;
  int mt = blockIdx.x * 8 + wave;             // 343*8 = 2744 tiles exactly
  const _Float16* arow = ctx + ((size_t)(mt * 16 + lane16)) * DIMD;
  v16h afrag[4];
  for (int kc = 0; kc < 4; ++kc) {
    const _Float16* ap = arow + kc * 32 + half * 16;
    afrag[kc] = ld16(ap, ap + 8);
  }
  for (int nt = 0; nt < 8; ++nt) {
    v8f acc = {};
    const _Float16* bcol = woutT + (nt * 16 + lane16) * DIMD;
    for (int kc = 0; kc < 4; ++kc) {
      const _Float16* bp = bcol + kc * 32 + half * 16;
      acc = wmma32(afrag[kc], ld16(bp, bp + 8), acc);
    }
    for (int r = 0; r < 8; ++r) {
      int row = mt * 16 + r + half * 8;
      out[(size_t)row * DIMD + nt * 16 + lane16] = acc[r];
    }
  }
}

// ---------------- launch ----------------
extern "C" void kernel_launch(void* const* d_in, const int* in_sizes, int n_in,
                              void* d_out, int out_size, void* d_ws, size_t ws_size,
                              hipStream_t stream) {
  const float* x    = (const float*)d_in[0];
  const float* lnw  = (const float*)d_in[1];
  const float* lnb  = (const float*)d_in[2];
  const float* wqkv = (const float*)d_in[3];
  const float* wout = (const float*)d_in[4];
  const float* btab = (const float*)d_in[5];
  const int*   rel  = (const int*)d_in[6];

  char* ws = (char*)d_ws;   // requires ~45 MB scratch
  _Float16* wqkvT = (_Float16*)(ws + OFF_WQKVT);
  _Float16* woutT = (_Float16*)(ws + OFF_WOUTT);
  float*    biasm = (float*)(ws + OFF_BIAS);
  _Float16* qws   = (_Float16*)(ws + OFF_Q);
  _Float16* kws   = (_Float16*)(ws + OFF_K);
  _Float16* vws   = (_Float16*)(ws + OFF_V);
  _Float16* ctx   = (_Float16*)(ws + OFF_CTX);

  hipLaunchKernelGGL(prep_weights, dim3(256), dim3(256), 0, stream, wqkv, wout, wqkvT, woutT);
  hipLaunchKernelGGL(prep_bias, dim3((HEADS_ * NTOK * NTOK + 255) / 256), dim3(256), 0, stream,
                     btab, rel, biasm);
  hipLaunchKernelGGL(ln_qkv, dim3(NWIN, 3), dim3(256), 0, stream, x, lnw, lnb, wqkvT, qws, kws, vws);
  hipLaunchKernelGGL(attn, dim3(NWIN * HEADS_), dim3(256), 0, stream, qws, kws, vws, biasm, ctx);
  hipLaunchKernelGGL(proj, dim3(343), dim3(256), 0, stream, ctx, woutT, (float*)d_out);
}
